// LanguageModel_41609643164460
// MI455X (gfx1250) — compile-verified, matrix operand order checked by reading, not run on hardware
//
#include <hip/hip_runtime.h>
#include <math.h>
#include <stdint.h>

typedef __attribute__((ext_vector_type(16))) __bf16 v16bf;
typedef __attribute__((ext_vector_type(8)))  __bf16 v8bf;
typedef __attribute__((ext_vector_type(8)))  float  v8f;

#define NSB 8   // vocab slice blocks (grid.y of gemm)
#define NW  4   // waves per gemm block (blockDim = 128 = 4 * wave32)
#define NT  4   // N-tiles (16 cols each) register-blocked per wave iteration

// ---------------------------------------------------------------------------
// 1) Build packed fwd/bwd (context, target-token) rows from batch_sizes.
// ---------------------------------------------------------------------------
__global__ void setup_kernel(const int* __restrict__ bs, const int* __restrict__ sent,
                             int T, int maxrows,
                             int* __restrict__ meta, int* __restrict__ src,
                             int* __restrict__ token, float* __restrict__ hdr) {
    if (threadIdx.x != 0 || blockIdx.x != 0) return;
    if (T > 256) T = 256;
    int off[257];
    off[0] = 0;
    for (int t = 0; t < T; ++t) off[t + 1] = off[t] + bs[t];
    int total = off[T];
    int row = 0;
    // forward: hidden at t-1 predicts token at t
    for (int t = 1; t < T; ++t)
        for (int j = 0; j < bs[t] && row < maxrows; ++j) {
            meta[row] = 1; src[row] = off[t - 1] + j; token[row] = sent[off[t] + j]; ++row;
        }
    while ((row & 15) && row < maxrows) { meta[row] = 0; src[row] = 0; token[row] = 0; ++row; }
    // backward: hidden at i+1 predicts token at i
    for (int i = 1; i < T - 1; ++i)
        for (int j = 0; j < bs[i + 1] && row < maxrows; ++j) {
            meta[row] = 2; src[row] = off[i + 1] + j; token[row] = sent[off[i] + j]; ++row;
        }
    while (row < maxrows) { meta[row] = 0; src[row] = 0; token[row] = 0; ++row; }
    hdr[0] = (float)(2 * total - bs[0] - bs[T - 1]);   // denom
}

// ---------------------------------------------------------------------------
// 2) Gather context rows of hidden_states (fwd: [:,0:D], bwd: [:,D:2D]) -> bf16 A
// ---------------------------------------------------------------------------
__global__ void gather_a(const float* __restrict__ hs, const int* __restrict__ meta,
                         const int* __restrict__ src, int DIM, __bf16* __restrict__ A) {
    int row = blockIdx.x;
    int m = meta[row];
    const float* h = hs + (size_t)src[row] * (2 * DIM) + (m == 2 ? DIM : 0);
    for (int i = threadIdx.x; i < DIM; i += blockDim.x) {
        float v = (m == 0) ? 0.f : h[i];
        A[(size_t)row * DIM + i] = (__bf16)v;
    }
}

// ---------------------------------------------------------------------------
// 3) Convert the classifier weight to bf16 once (65 MB -> L2-resident).
// ---------------------------------------------------------------------------
__global__ void conv_w(const float* __restrict__ W, int DIM, __bf16* __restrict__ Wb) {
    size_t r = blockIdx.x;
    for (int i = threadIdx.x; i < DIM; i += blockDim.x)
        Wb[r * DIM + i] = (__bf16)W[r * DIM + i];
}

// ---------------------------------------------------------------------------
// 4) Gold logit per row, full fp32: dot(h, W[token]) + bias[token].
// ---------------------------------------------------------------------------
__global__ void gold_kernel(const float* __restrict__ hs, const float* __restrict__ W,
                            const float* __restrict__ fb, const float* __restrict__ bb,
                            const int* __restrict__ meta, const int* __restrict__ src,
                            const int* __restrict__ token, int DIM, float* __restrict__ gold) {
    __shared__ float red[256];
    int row = blockIdx.x;
    int m = meta[row];
    if (m == 0) { if (threadIdx.x == 0) gold[row] = 0.f; return; }
    const float* h = hs + (size_t)src[row] * (2 * DIM) + (m == 2 ? DIM : 0);
    const float* w = W + (size_t)token[row] * DIM;
    float s = 0.f;
    for (int i = threadIdx.x; i < DIM; i += blockDim.x) s += h[i] * w[i];
    red[threadIdx.x] = s; __syncthreads();
    for (int st = 128; st > 0; st >>= 1) {
        if ((int)threadIdx.x < st) red[threadIdx.x] += red[threadIdx.x + st];
        __syncthreads();
    }
    if (threadIdx.x == 0)
        gold[row] = red[0] + ((m == 2) ? bb[token[row]] : fb[token[row]]);
}

// ---------------------------------------------------------------------------
// 5) Fused bf16 WMMA GEMM + online logsumexp.
//    block = (16-row M-tile, 1/NSB of vocab); each wave register-blocks
//    NT=4 vocab tiles. All fragment loads of a K-chunk are hoisted ahead
//    of the 4 WMMAs so the scheduler can issue one load clause and use
//    staggered partial s_wait_loadcnt instead of full drains.
// ---------------------------------------------------------------------------
__global__ __launch_bounds__(128) void gemm_lse(
    const __bf16* __restrict__ A, const __bf16* __restrict__ Wb,
    const float* __restrict__ fb, const float* __restrict__ bb,
    const int* __restrict__ meta, int DIM, int VOCAB, int groupsPerBlk,
    float* __restrict__ pmax, float* __restrict__ psum) {

    int rowTile = blockIdx.x;
    int mseg = meta[rowTile * 16];
    if (mseg == 0) {
        // publish neutral partials so the reducer never reads poison
        if (threadIdx.x < 16) {
            size_t idx = (size_t)(rowTile * 16 + threadIdx.x) * gridDim.y + blockIdx.y;
            pmax[idx] = -INFINITY; psum[idx] = 0.f;
        }
        return;
    }
    const float* bias = (mseg == 2) ? bb : fb;

    int tid = threadIdx.x;
    int wave = tid >> 5, lane = tid & 31;
    int mcol = lane & 15, half = lane >> 4;
    int koff = half * 8;                       // 16-bit fragment half-split

    const __bf16* pA = A + (size_t)(rowTile * 16 + mcol) * DIM;

    int tilesTotal  = (VOCAB + 15) >> 4;
    int groupsTotal = (tilesTotal + NT - 1) / NT;
    int g0 = blockIdx.y * groupsPerBlk;
    int g1 = g0 + groupsPerBlk; if (g1 > groupsTotal) g1 = groupsTotal;

    float mx[8], sm[8];
#pragma unroll
    for (int r = 0; r < 8; ++r) { mx[r] = -INFINITY; sm[r] = 0.f; }

    for (int g = g0 + wave; g < g1; g += NW) {
        int colBase = g * (NT * 16);
        int col[NT];
        const __bf16* pB[NT];
#pragma unroll
        for (int j = 0; j < NT; ++j) {
            col[j] = colBase + j * 16 + mcol;
            int cc = col[j] < VOCAB ? col[j] : VOCAB - 1;
            pB[j] = Wb + (size_t)cc * DIM;
        }
        if (g + NW < g1) {
            int nc = (g + NW) * (NT * 16) + mcol;
            if (nc >= VOCAB) nc = VOCAB - 1;
            __builtin_prefetch(Wb + (size_t)nc * DIM + koff, 0, 1);   // global_prefetch_b8
        }

        v8f acc[NT];
#pragma unroll
        for (int j = 0; j < NT; ++j) acc[j] = (v8f){};

#pragma unroll 2
        for (int kc = 0; kc < DIM; kc += 32) {
            // --- issue ALL loads for this K-chunk first (A + NT B fragments) ---
            const __bf16* a0 = pA + kc + koff;
            v8bf alo = *(const v8bf*)(a0);
            v8bf ahi = *(const v8bf*)(a0 + 16);
            v8bf blo[NT], bhi[NT];
#pragma unroll
            for (int j = 0; j < NT; ++j) {
                const __bf16* b0 = pB[j] + kc + koff;
                blo[j] = *(const v8bf*)(b0);
                bhi[j] = *(const v8bf*)(b0 + 16);
            }
            // --- then the NT dependent WMMAs (partial waits, loads overlap) ---
            v16bf af = __builtin_shufflevector(alo, ahi,
                           0,1,2,3,4,5,6,7,8,9,10,11,12,13,14,15);
#pragma unroll
            for (int j = 0; j < NT; ++j) {
                v16bf bg = __builtin_shufflevector(blo[j], bhi[j],
                               0,1,2,3,4,5,6,7,8,9,10,11,12,13,14,15);
                acc[j] = __builtin_amdgcn_wmma_f32_16x16x32_bf16(
                             false, af, false, bg, (short)0, acc[j], false, false);
            }
        }

#pragma unroll
        for (int j = 0; j < NT; ++j) {
            if (col[j] < VOCAB) {
                float b = bias[col[j]];
#pragma unroll
                for (int r = 0; r < 8; ++r) {
                    float x = acc[j][r] + b;
                    float mo = mx[r];
                    float mn = fmaxf(mo, x);
                    float e1 = (mo == mn) ? 1.f : __expf(mo - mn);
                    sm[r] = sm[r] * e1 + __expf(x - mn);
                    mx[r] = mn;
                }
            }
        }
    }

    // cross-lane (max,sum) merge within each 16-lane half:
    // row r lives in lanes 0-15, row r+8 in lanes 16-31 (C-fragment layout)
#pragma unroll
    for (int r = 0; r < 8; ++r) {
        float m1 = mx[r], s1 = sm[r];
#pragma unroll
        for (int d = 1; d < 16; d <<= 1) {
            float m2 = __shfl_xor(m1, d, 32);
            float s2 = __shfl_xor(s1, d, 32);
            float M = fmaxf(m1, m2);
            float e1 = (m1 == M) ? 1.f : __expf(m1 - M);
            float e2 = (m2 == M) ? 1.f : __expf(m2 - M);
            s1 = s1 * e1 + s2 * e2;
            m1 = M;
        }
        mx[r] = m1; sm[r] = s1;
    }

    __shared__ float lmax[NW][16], lsum[NW][16];
    if ((lane & 15) == 0) {
#pragma unroll
        for (int r = 0; r < 8; ++r) {
            lmax[wave][r + half * 8] = mx[r];
            lsum[wave][r + half * 8] = sm[r];
        }
    }
    __syncthreads();
    if (tid < 16) {
        float M = lmax[0][tid], S = lsum[0][tid];
        for (int w = 1; w < NW; ++w) {
            float m2 = lmax[w][tid], s2 = lsum[w][tid];
            float Mn = fmaxf(M, m2);
            float e1 = (M == Mn) ? 1.f : __expf(M - Mn);
            float e2 = (m2 == Mn) ? 1.f : __expf(m2 - Mn);
            S = S * e1 + s2 * e2; M = Mn;
        }
        size_t idx = (size_t)(rowTile * 16 + tid) * gridDim.y + blockIdx.y;
        pmax[idx] = M; psum[idx] = S;
    }
}

// ---------------------------------------------------------------------------
// 6) Merge vocab-slice partials -> lse, subtract gold, block-sum.
// ---------------------------------------------------------------------------
__global__ void reduce_rows(const int* __restrict__ meta, const float* __restrict__ gold,
                            const float* __restrict__ pmax, const float* __restrict__ psum,
                            int nsb, int maxrows, float* __restrict__ bsum) {
    __shared__ float red[256];
    int row = blockIdx.x * 256 + threadIdx.x;
    float c = 0.f;
    if (row < maxrows && meta[row] != 0) {
        float M = -INFINITY, S = 0.f;
        for (int s = 0; s < nsb; ++s) {
            float m2 = pmax[(size_t)row * nsb + s], s2 = psum[(size_t)row * nsb + s];
            float Mn = fmaxf(M, m2);
            float e1 = (M == Mn) ? 1.f : __expf(M - Mn);
            float e2 = (m2 == Mn) ? 1.f : __expf(m2 - Mn);
            S = S * e1 + s2 * e2; M = Mn;
        }
        c = M + __logf(S) - gold[row];
    }
    red[threadIdx.x] = c; __syncthreads();
    for (int st = 128; st > 0; st >>= 1) {
        if ((int)threadIdx.x < st) red[threadIdx.x] += red[threadIdx.x + st];
        __syncthreads();
    }
    if (threadIdx.x == 0) bsum[blockIdx.x] = red[0];
}

__global__ void finalize(const float* __restrict__ bsum, int nblk,
                         const float* __restrict__ hdr, float* __restrict__ out) {
    if (threadIdx.x != 0 || blockIdx.x != 0) return;
    float s = 0.f;
    for (int i = 0; i < nblk; ++i) s += bsum[i];
    out[0] = s / hdr[0];
}

// ---------------------------------------------------------------------------
extern "C" void kernel_launch(void* const* d_in, const int* in_sizes, int n_in,
                              void* d_out, int out_size, void* d_ws, size_t ws_size,
                              hipStream_t stream) {
    const float* hs   = (const float*)d_in[0];   // [total, 2*DIM]
    const float* W    = (const float*)d_in[1];   // [VOCAB, DIM]
    const float* fb   = (const float*)d_in[2];   // [VOCAB]
    const float* bb   = (const float*)d_in[3];   // [VOCAB]
    const int*   sent = (const int*)d_in[4];     // [total]
    const int*   bs   = (const int*)d_in[5];     // [T]

    int total = in_sizes[4];
    int T     = in_sizes[5];
    int VOCAB = in_sizes[2];
    int DIM   = in_sizes[1] / VOCAB;
    int maxrows = (((2 * total + 15) / 16) * 16) + 32;  // fwd + bwd rows + tile padding

    // workspace carve-out (256B aligned)
    uint8_t* w = (uint8_t*)d_ws; size_t off = 0;
    auto alloc = [&](size_t b) -> void* {
        void* p = w + off; off = (off + b + 255) & ~(size_t)255; return p;
    };
    float*  hdr   = (float*)alloc(256);
    int*    meta  = (int*)alloc(sizeof(int) * maxrows);
    int*    src   = (int*)alloc(sizeof(int) * maxrows);
    int*    token = (int*)alloc(sizeof(int) * maxrows);
    float*  gold  = (float*)alloc(sizeof(float) * maxrows);
    float*  pmax  = (float*)alloc(sizeof(float) * (size_t)maxrows * NSB);
    float*  psum  = (float*)alloc(sizeof(float) * (size_t)maxrows * NSB);
    float*  bsum  = (float*)alloc(sizeof(float) * 64);
    __bf16* A     = (__bf16*)alloc(sizeof(__bf16) * (size_t)maxrows * DIM);
    __bf16* Wb    = (__bf16*)alloc(sizeof(__bf16) * (size_t)VOCAB * DIM);
    (void)ws_size; (void)n_in; (void)out_size;

    setup_kernel<<<1, 1, 0, stream>>>(bs, sent, T, maxrows, meta, src, token, hdr);
    gather_a<<<maxrows, 256, 0, stream>>>(hs, meta, src, DIM, A);
    conv_w<<<VOCAB, 256, 0, stream>>>(W, DIM, Wb);
    gold_kernel<<<maxrows, 256, 0, stream>>>(hs, W, fb, bb, meta, src, token, DIM, gold);

    int tilesTotal   = (VOCAB + 15) / 16;
    int groupsTotal  = (tilesTotal + NT - 1) / NT;
    int groupsPerBlk = (groupsTotal + NSB - 1) / NSB;
    gemm_lse<<<dim3(maxrows / 16, NSB), 128, 0, stream>>>(
        A, Wb, fb, bb, meta, DIM, VOCAB, groupsPerBlk, pmax, psum);

    int nblk = (maxrows + 255) / 256;
    reduce_rows<<<nblk, 256, 0, stream>>>(meta, gold, pmax, psum, NSB, maxrows, bsum);
    finalize<<<1, 1, 0, stream>>>(bsum, nblk, hdr, (float*)d_out);
}
